// RNN_chaotic_51350628991153
// MI455X (gfx1250) — matrix operand order repaired and checked by reading, not compiled, not run on hardware
//
#include <hip/hip_runtime.h>
#include <hip/hip_bf16.h>
#include <stdint.h>

// ---------------- problem constants ----------------
#define TT      1024
#define BB      32
#define II      64
#define HH      1024
#define OFRQ    64
#define OCTX    8
#define AL      0.1f

#define NWG     16          // persistent workgroups for the scan
#define SLICE   64          // HH / NWG hidden columns per WG
#define THREADS 256         // 8 waves (wave32)

// ---------------- WMMA vector types ----------------
typedef __attribute__((ext_vector_type(16))) __bf16 v16bf;
typedef __attribute__((ext_vector_type(8)))  float  v8f;

// ---------------- LDS layout (recurrence) ----------
// strides in elements; row stride bytes (2064/144) stay 16B-aligned for b128 loads
#define WH_STR  1032
#define RL_STR  1032
#define WI_STR  72
#define XI_STR  72
#define RS_STR  68
#define WH_OFF  0
#define RL_OFF  (WH_OFF + SLICE*WH_STR*2)     // 132096
#define WI_OFF  (RL_OFF + BB*RL_STR*2)        // 198144
#define XI_OFF  (WI_OFF + SLICE*WI_STR*2)     // 207360
#define RS_OFF  (XI_OFF + BB*XI_STR*2)        // 211968
#define LDS_RECUR_BYTES (RS_OFF + BB*RS_STR*4) // 220672  (< 320KB WGP LDS)

// ---------------- LDS layout (readout) --------------
#define WO_STR  1032
#define WC_STR  1032
#define RA_STR  1032
#define WO_OFF  0
#define WC_OFF  (WO_OFF + OFRQ*WO_STR*2)      // 132096
#define RA_OFF  (WC_OFF + 16*WC_STR*2)        // 165120
#define LDS_READ_BYTES (RA_OFF + BB*RA_STR*2) // 231168

// ---------------- workspace layout ------------------
#define WS_CNT  0                              // u32 barrier counter
#define WS_BUF0 4096                           // bf16 r state ping [32][1024]
#define WS_BUF1 (4096 + BB*HH*2)               // bf16 r state pong

// ---------------- helpers ---------------------------
__device__ __forceinline__ unsigned short f2bf(float f) {
  union { float f; unsigned u; } c; c.f = f;
  unsigned r = 0x7FFFu + ((c.u >> 16) & 1u);   // round-to-nearest-even
  return (unsigned short)((c.u + r) >> 16);
}
__device__ __forceinline__ unsigned pack2(float a, float b) {
  return (unsigned)f2bf(a) | ((unsigned)f2bf(b) << 16);
}

// A-operand (16x32 bf16). Per the CDNA5 A layout, lane (half,m) holds K pairs
// p(v) = (v>=4)*16 + half*8 + (v&3)*2  -> two contiguous 16B runs per lane.
__device__ __forceinline__ v16bf load_a(const unsigned short* base, int stride,
                                        int row0, int lane, int kbase) {
  const int half = lane >> 4, m = lane & 15;
  const unsigned short* rp = base + (size_t)(row0 + m) * stride + kbase + half * 8;
  const uint4 lo = *(const uint4*)(rp);        // ds_load_b128
  const uint4 hi = *(const uint4*)(rp + 16);   // ds_load_b128
  unsigned u[8] = { lo.x, lo.y, lo.z, lo.w, hi.x, hi.y, hi.z, hi.w };
  v16bf out; __builtin_memcpy(&out, u, sizeof(u)); return out;
}

// B-operand (32x16 bf16) from row-major W[j][k] (gives W^T): lane n = column j;
// same K pairing as load_a: half*8 + {0..7} and 16 + half*8 + {0..7} -> two 16B runs/lane.
__device__ __forceinline__ v16bf load_b(const unsigned short* base, int stride,
                                        int col0, int lane, int kbase) {
  const int half = lane >> 4, n = lane & 15;
  const unsigned short* rp = base + (size_t)(col0 + n) * stride + kbase + half * 8;
  const uint4 lo = *(const uint4*)(rp);        // ds_load_b128
  const uint4 hi = *(const uint4*)(rp + 16);   // ds_load_b128
  unsigned u[8] = { lo.x, lo.y, lo.z, lo.w, hi.x, hi.y, hi.z, hi.w };
  v16bf out; __builtin_memcpy(&out, u, sizeof(u)); return out;
}

#define WMMA_BF16(A, B, C) \
  __builtin_amdgcn_wmma_f32_16x16x32_bf16(false, (A), false, (B), (short)0, (C), false, false)

// Software-pipelined K-reduction over [klo, khi) chunks of 32:
// next chunk's ds_load_b128s are issued before the current WMMA so the
// scheduler can wait on dscnt<=4 instead of draining to zero each chunk.
__device__ __forceinline__ v8f kloop(const unsigned short* abase, int astr, int arow,
                                     const unsigned short* bbase, int bstr, int bcol,
                                     int lane, int nchunk, v8f acc) {
  v16bf a_cur = load_a(abase, astr, arow, lane, 0);
  v16bf b_cur = load_b(bbase, bstr, bcol, lane, 0);
#pragma unroll 4
  for (int kc = 0; kc < nchunk - 1; ++kc) {
    v16bf a_nxt = load_a(abase, astr, arow, lane, (kc + 1) * 32);
    v16bf b_nxt = load_b(bbase, bstr, bcol, lane, (kc + 1) * 32);
    acc = WMMA_BF16(a_cur, b_cur, acc);
    a_cur = a_nxt; b_cur = b_nxt;
  }
  return WMMA_BF16(a_cur, b_cur, acc);
}

// monotonic-counter grid barrier (counter zeroed by hipMemsetAsync each launch)
__device__ __forceinline__ void gbar(unsigned* cnt, unsigned target) {
  __syncthreads();
  if (threadIdx.x == 0) {
    __atomic_fetch_add(cnt, 1u, __ATOMIC_RELEASE);
    while (__atomic_load_n(cnt, __ATOMIC_ACQUIRE) < target)
      __builtin_amdgcn_s_sleep(2);
  }
  __syncthreads();
}

// ================= recurrence kernel =================
__global__ void rnn_recur_kernel(const float* __restrict__ x,     // [T,B,I]
                                 const float* __restrict__ r0,    // [B,H]
                                 const float* __restrict__ wi,    // [H,I]
                                 const float* __restrict__ bi,    // [H]
                                 const float* __restrict__ wh,    // [H,H]
                                 const float* __restrict__ bh,    // [H]
                                 float* __restrict__ rate_out,    // [T,B,H] (inside d_out)
                                 unsigned short* __restrict__ buf0,
                                 unsigned short* __restrict__ buf1,
                                 unsigned* __restrict__ cnt) {
  extern __shared__ char smem[];
  unsigned short* WH = (unsigned short*)(smem + WH_OFF);
  unsigned short* RL = (unsigned short*)(smem + RL_OFF);
  unsigned short* WI = (unsigned short*)(smem + WI_OFF);
  unsigned short* XI = (unsigned short*)(smem + XI_OFF);
  float*          RS = (float*)(smem + RS_OFF);

  const int tid  = threadIdx.x;
  const int wg   = blockIdx.x;
  const int lane = tid & 31, wid = tid >> 5;
  const int mt   = wid >> 2, nt = wid & 3;     // 2 x 4 tile grid, 1 tile/wave

  // -------- prologue: resident weights -> LDS bf16 (done once) --------
  for (int p = tid; p < SLICE * HH / 2; p += THREADS) {      // h2h slice [64][1024]
    const int j = p >> 9, kp = p & 511;
    const float* s = wh + (size_t)(wg * SLICE + j) * HH + kp * 2;
    *(unsigned*)(WH + j * WH_STR + kp * 2) = pack2(s[0], s[1]);
  }
  for (int p = tid; p < SLICE * II / 2; p += THREADS) {      // i2h slice [64][64]
    const int j = p >> 5, kp = p & 31;
    const float* s = wi + (size_t)(wg * SLICE + j) * II + kp * 2;
    *(unsigned*)(WI + j * WI_STR + kp * 2) = pack2(s[0], s[1]);
  }
  for (int e = tid; e < BB * SLICE; e += THREADS) {          // fp32 state slice + bf16 seed
    const int b = e >> 6, jl = e & 63;
    const float v = r0[(size_t)b * HH + wg * SLICE + jl];
    RS[b * RS_STR + jl] = v;
    buf0[(size_t)b * HH + wg * SLICE + jl] = f2bf(v);
  }
  const int jcol = wg * SLICE + nt * 16 + (lane & 15);
  const float bias = bi[jcol] + bh[jcol];

  unsigned bar_target = NWG;
  __threadfence();
  gbar(cnt, bar_target);                                     // all seeds visible

  const unsigned lds_rl = (unsigned)(uintptr_t)RL;           // LDS offset for async dst

  for (int t = 0; t < TT; ++t) {
    const unsigned short* rbuf = (t & 1) ? buf1 : buf0;
    unsigned short*       wbuf = (t & 1) ? buf0 : buf1;

    // ---- async-copy full bf16 state [32][1024] -> LDS (CDNA5 path) ----
    for (int c = tid; c < (BB * HH * 2) / 16; c += THREADS) { // 4096 x 16B
      const int row = c >> 7, c16 = c & 127;
      const char* gp = (const char*)rbuf + row * 2048 + c16 * 16;
      const unsigned loff = lds_rl + row * (RL_STR * 2) + c16 * 16;
      asm volatile("global_load_async_to_lds_b128 %0, %1, off"
                   :: "v"(loff), "v"(gp) : "memory");
    }
    // ---- x_t [32][64] fp32 -> bf16 LDS (fused input projection) ----
    {
      const float* xt = x + (size_t)t * BB * II;
      for (int p = tid; p < BB * II / 2; p += THREADS) {
        const int b = p >> 5, kp = p & 31;
        const float* s = xt + b * II + kp * 2;
        *(unsigned*)(XI + b * XI_STR + kp * 2) = pack2(s[0], s[1]);
      }
    }
    asm volatile("s_wait_asynccnt 0x0" ::: "memory");
    __syncthreads();

    // ---- one 16x16 output tile per wave: K = 64 (input) + 1024 (recurrent) ----
    v8f acc = {};
    acc = kloop(XI, XI_STR, mt * 16, WI, WI_STR, nt * 16, lane, II / 32, acc);
    acc = kloop(RL, RL_STR, mt * 16, WH, WH_STR, nt * 16, lane, HH / 32, acc);

    // ---- leaky tanh update (fp32 state kept in LDS), emit outputs ----
    const int col = lane & 15, rb = (lane >> 4) * 8, jl = nt * 16 + col;
    float* rate_t = rate_out + (size_t)t * BB * HH;
#pragma unroll
    for (int i = 0; i < 8; ++i) {
      const int row = mt * 16 + rb + i;
      const float pre  = acc[i] + bias;
      const float rold = RS[row * RS_STR + jl];
      const float rn   = (1.0f - AL) * rold + AL * tanhf(pre);
      RS[row * RS_STR + jl] = rn;
      rate_t[(size_t)row * HH + wg * SLICE + jl] = rn;          // fp32 output
      wbuf[(size_t)row * HH + wg * SLICE + jl]   = f2bf(rn);    // bf16 next-step state
    }

    __threadfence();
    bar_target += NWG;
    gbar(cnt, bar_target);
  }
}

// ================= readout kernel =================
__global__ void rnn_readout_kernel(const float* __restrict__ rate_all, // [T,B,H]
                                   const float* __restrict__ wo,       // [64,H]
                                   const float* __restrict__ bo,
                                   const float* __restrict__ wc,       // [8,H]
                                   const float* __restrict__ bc,
                                   float* __restrict__ out_f,          // [T,B,64]
                                   float* __restrict__ out_c) {        // [T,B,8]
  extern __shared__ char smem[];
  unsigned short* WO = (unsigned short*)(smem + WO_OFF);
  unsigned short* WC = (unsigned short*)(smem + WC_OFF);
  unsigned short* RA = (unsigned short*)(smem + RA_OFF);

  const int tid = threadIdx.x, lane = tid & 31, wid = tid >> 5;

  for (int p = tid; p < OFRQ * HH / 2; p += THREADS) {
    const int j = p >> 9, kp = p & 511;
    const float* s = wo + (size_t)j * HH + kp * 2;
    *(unsigned*)(WO + j * WO_STR + kp * 2) = pack2(s[0], s[1]);
  }
  for (int p = tid; p < 16 * HH / 2; p += THREADS) {   // ctx padded to 16 rows
    const int j = p >> 9, kp = p & 511;
    unsigned u = 0;
    if (j < OCTX) { const float* s = wc + (size_t)j * HH + kp * 2; u = pack2(s[0], s[1]); }
    *(unsigned*)(WC + j * WC_STR + kp * 2) = u;
  }

  for (int t = blockIdx.x; t < TT; t += gridDim.x) {
    __syncthreads();
    const float* rt = rate_all + (size_t)t * BB * HH;
    for (int p = tid; p < BB * HH / 2; p += THREADS) {
      const int b = p >> 9, kp = p & 511;
      const float* s = rt + (size_t)b * HH + kp * 2;
      *(unsigned*)(RA + b * RA_STR + kp * 2) = pack2(s[0], s[1]);
    }
    __syncthreads();

    const int col = lane & 15, rb = (lane >> 4) * 8;
    {   // frequency readout: 8 tiles, one per wave
      const int mt = wid >> 2, nt = wid & 3;
      v8f acc = {};
      acc = kloop(RA, RA_STR, mt * 16, WO, WO_STR, nt * 16, lane, HH / 32, acc);
      const float bias = bo[nt * 16 + col];
#pragma unroll
      for (int i = 0; i < 8; ++i) {
        const int row = mt * 16 + rb + i;
        out_f[(size_t)t * BB * OFRQ + row * OFRQ + nt * 16 + col] = acc[i] + bias;
      }
    }
    if (wid < 2) {  // context readout (N padded 8->16; predicate only the store)
      const int mt = wid;
      v8f acc = {};
      acc = kloop(RA, RA_STR, mt * 16, WC, WC_STR, 0, lane, HH / 32, acc);
      if (col < OCTX) {
        const float bias = bc[col];
#pragma unroll
        for (int i = 0; i < 8; ++i) {
          const int row = mt * 16 + rb + i;
          out_c[(size_t)t * BB * OCTX + row * OCTX + col] = acc[i] + bias;
        }
      }
    }
  }
}

// ================= host launcher =================
extern "C" void kernel_launch(void* const* d_in, const int* in_sizes, int n_in,
                              void* d_out, int out_size, void* d_ws, size_t ws_size,
                              hipStream_t stream) {
  (void)in_sizes; (void)n_in; (void)out_size; (void)ws_size;
  const float* input_sig = (const float*)d_in[0];
  const float* rate      = (const float*)d_in[1];
  const float* i2h_w     = (const float*)d_in[2];
  const float* i2h_b     = (const float*)d_in[3];
  const float* h2h_w     = (const float*)d_in[4];
  const float* h2h_b     = (const float*)d_in[5];
  const float* h2o_w     = (const float*)d_in[6];
  const float* h2o_b     = (const float*)d_in[7];
  const float* h2o_ctx_w = (const float*)d_in[8];
  const float* h2o_ctx_b = (const float*)d_in[9];

  float* out_f  = (float*)d_out;                                   // [T,B,64]
  float* out_c  = out_f + (size_t)TT * BB * OFRQ;                  // [T,B,8]
  float* rate_a = out_f + (size_t)TT * BB * (OFRQ + OCTX);         // [T,B,H]

  unsigned*       cnt  = (unsigned*)((char*)d_ws + WS_CNT);
  unsigned short* buf0 = (unsigned short*)((char*)d_ws + WS_BUF0);
  unsigned short* buf1 = (unsigned short*)((char*)d_ws + WS_BUF1);

  // reset barrier counter every call -> deterministic across graph replays
  hipMemsetAsync(d_ws, 0, 4096, stream);

  hipFuncSetAttribute((const void*)rnn_recur_kernel,
                      hipFuncAttributeMaxDynamicSharedMemorySize, LDS_RECUR_BYTES);
  hipFuncSetAttribute((const void*)rnn_readout_kernel,
                      hipFuncAttributeMaxDynamicSharedMemorySize, LDS_READ_BYTES);

  rnn_recur_kernel<<<NWG, THREADS, LDS_RECUR_BYTES, stream>>>(
      input_sig, rate, i2h_w, i2h_b, h2h_w, h2h_b, rate_a, buf0, buf1, cnt);

  rnn_readout_kernel<<<128, THREADS, LDS_READ_BYTES, stream>>>(
      rate_a, h2o_w, h2o_b, h2o_ctx_w, h2o_ctx_b, out_f, out_c);
}
